// RGCNModel_30073361007327
// MI455X (gfx1250) — matrix operand order repaired and checked by reading, not compiled, not run on hardware
//
#include <hip/hip_runtime.h>
#include <hip/hip_bf16.h>
#include <stdint.h>

// RGCN (basis decomposition) restructured for MI455X / gfx1250:
//  - per-edge matmuls hoisted to node-level bf16 WMMA GEMMs (16x16x32, f32 acc)
//  - basis-product tensor Y kept in bf16 (L2-resident: ~54MB total << 192MB L2)
//  - A-tiles staged to LDS via GLOBAL_LOAD_ASYNC_TO_LDS_B128 (ASYNCcnt path)
//  - f32 accumulation for segment-sums / softmax

typedef __attribute__((ext_vector_type(16))) __bf16 v16bf;
typedef __attribute__((ext_vector_type(8)))  float  v8f;

// exact parameter types for the async-to-LDS builtin (from hipcc diagnostic):
// param0 = int __attribute__((vector_size(16))) in addrspace(1)
typedef int vsi4 __attribute__((vector_size(16)));
typedef __attribute__((address_space(1))) vsi4* gp4_t;   // global v4i*
typedef __attribute__((address_space(3))) vsi4* lp4_t;   // LDS    v4i*

#define HDIM  128
#define BASES 4

#if defined(__HIP_DEVICE_COMPILE__) && __has_builtin(__builtin_amdgcn_global_load_async_to_lds_b128)
#define HAVE_ASYNC_LDS 1
#else
#define HAVE_ASYNC_LDS 0
#endif

__device__ __forceinline__ void wait_asynccnt0() {
#if defined(__HIP_DEVICE_COMPILE__)
#if __has_builtin(__builtin_amdgcn_s_wait_asynccnt)
  __builtin_amdgcn_s_wait_asynccnt(0);
#else
  asm volatile("s_wait_asynccnt 0" ::: "memory");
#endif
#endif
}

// ---------------- zero ----------------
__global__ void k_zero_f32(float* __restrict__ p, int n4) {
  int i = blockIdx.x * blockDim.x + threadIdx.x;
  if (i < n4) ((float4*)p)[i] = make_float4(0.f, 0.f, 0.f, 0.f);
}

// ---------------- edge gather / combine / scatter ----------------
struct bf16x4 { __bf16 a, b, c, d; };

__device__ __forceinline__ void ld4(const float* p, float v[4]) {
  float4 t = *(const float4*)p;
  v[0] = t.x; v[1] = t.y; v[2] = t.z; v[3] = t.w;
}
__device__ __forceinline__ void ld4(const __bf16* p, float v[4]) {
  bf16x4 t = *(const bf16x4*)p;
  v[0] = (float)t.a; v[1] = (float)t.b; v[2] = (float)t.c; v[3] = (float)t.d;
}

// one wave per edge; lane handles 4 channels (32 lanes x 4 = 128)
template <typename T>
__global__ void __launch_bounds__(256) k_edge_combine(
    const int* __restrict__ src, const int* __restrict__ dst,
    const int* __restrict__ rel, const float* __restrict__ norm,
    const float* __restrict__ comp,          // [R,4]
    const T* __restrict__ Y, long long strideN, long long strideB,
    const int* __restrict__ xmap,            // non-null only for input layer
    float* __restrict__ hout, int E)
{
  int wid = threadIdx.x >> 5, lane = threadIdx.x & 31;
  int e = blockIdx.x * 8 + wid;
  if (e >= E) return;
  int s = src[e];
  if (xmap) s = xmap[s];
  int d = dst[e];
  int r = rel[e];
  float nm = norm[e];
  float c0 = comp[r*4+0], c1 = comp[r*4+1], c2 = comp[r*4+2], c3 = comp[r*4+3];
  int ch = lane * 4;
  const T* base = Y + (long long)s * strideN + ch;
  float v0[4], v1[4], v2[4], v3[4];
  ld4(base,             v0);
  ld4(base +   strideB, v1);
  ld4(base + 2*strideB, v2);
  ld4(base + 3*strideB, v3);
  float* out = hout + (long long)d * HDIM + ch;
  #pragma unroll
  for (int j = 0; j < 4; ++j) {
    float m = nm * (c0*v0[j] + c1*v1[j] + c2*v2[j] + c3*v3[j]);
    atomicAdd(&out[j], m);
  }
}

// ---------------- bias + relu + bf16 snapshot ----------------
__global__ void k_finalize(float* __restrict__ h, const float* __restrict__ bias,
                           __bf16* __restrict__ obf, int total, int do_relu)
{
  int i = blockIdx.x * blockDim.x + threadIdx.x;
  if (i >= total) return;
  float v = h[i] + bias[i & (HDIM-1)];
  if (do_relu) v = fmaxf(v, 0.f);
  h[i] = v;
  if (obf) obf[i] = (__bf16)v;
}

// ---------------- pack V[b,k,o] (f32) -> fragment-ready bf16 B ----------------
// packed index p = ((ct*4 + kc)*32 + lane)*16 + e  maps to
//   k = kc*32 + (lane>>4)*16 + e ,  ncol = ct*16 + (lane&15), b=ncol>>7, o=ncol&127
__global__ void k_pack_w(const float* __restrict__ V, __bf16* __restrict__ Wpk, int total)
{
  int p = blockIdx.x * blockDim.x + threadIdx.x;
  if (p >= total) return;
  int e  = p & 15;
  int l  = (p >> 4) & 31;
  int kc = (p >> 9) & 3;
  int ct = p >> 11;
  int k    = kc*32 + (l >> 4)*16 + e;
  int ncol = ct*16 + (l & 15);
  int b = ncol >> 7, o = ncol & 127;
  Wpk[p] = (__bf16) V[ (size_t)(b*HDIM + k)*HDIM + o ];
}

// ---------------- bf16 WMMA GEMM: Y[N,512] = A[N,128] @ W[128,512] ----------------
__global__ void __launch_bounds__(256) k_gemm_wmma(
    const __bf16* __restrict__ A,    // [N,128] row-major bf16
    const __bf16* __restrict__ Bpk,  // fragment-packed [32 ct][4 kc][32 lane][16]
    __bf16* __restrict__ Y,          // [N,512] row-major bf16
    int nrows)
{
  __shared__ __bf16 As[16 * HDIM];   // 16 rows x 128 K, 4KB
  const int rowbase = blockIdx.x * 16;
  const int tid = threadIdx.x;
  const bool fullTile = (rowbase + 16 <= nrows);  // wave-uniform

  { // stage A tile: 256 threads x 16B (8 bf16) = 2048 elems
    int r   = tid >> 4;
    int seg = (tid & 15) * 8;
    int gr  = rowbase + r;
    if (gr < nrows) {
#if HAVE_ASYNC_LDS
      __builtin_amdgcn_global_load_async_to_lds_b128(
          (gp4_t)&A[(size_t)gr*HDIM + seg],
          (lp4_t)&As[r*HDIM + seg],
          /*offset=*/0, /*cpol=*/0);
#else
      *(uint4*)&As[r*HDIM + seg] = *(const uint4*)&A[(size_t)gr*HDIM + seg];
#endif
    } else {
      *(uint4*)&As[r*HDIM + seg] = make_uint4(0, 0, 0, 0);
    }
  }
#if HAVE_ASYNC_LDS
  wait_asynccnt0();
#endif
  __syncthreads();

  const int wid = tid >> 5, lane = tid & 31;
  const int ct = blockIdx.y * 8 + wid;       // column tile 0..31 (16 cols each)
  const int half = lane >> 4, mr = lane & 15;

  v8f c = {0.f,0.f,0.f,0.f,0.f,0.f,0.f,0.f};
  #pragma unroll
  for (int kc = 0; kc < 4; ++kc) {
    v16bf a, b;
    // A fragment: lane half owns contiguous K {h*16..h*16+15} (matches packed B)
    const __bf16* arow = &As[mr*HDIM + kc*32];
    ((uint4*)&a)[0] = *(const uint4*)&arow[half*16];
    ((uint4*)&a)[1] = *(const uint4*)&arow[half*16 + 8];
    // B fragment: pre-swizzled, contiguous 32B per lane
    const uint4* bp = (const uint4*)&Bpk[(((size_t)ct*4 + kc)*32 + lane)*16];
    ((uint4*)&b)[0] = bp[0];
    ((uint4*)&b)[1] = bp[1];
    c = __builtin_amdgcn_wmma_f32_16x16x32_bf16(
            /*neg_a=*/false, a, /*neg_b=*/false, b,
            /*c_mod=*/(short)0, c, /*reuse_a=*/false, /*reuse_b=*/false);
  }

  // D layout: lanes 0-15 -> M=g, lanes 16-31 -> M=g+8 ; N = lane&15
  const int col = ct*16 + mr;
  const int m0  = rowbase + half*8;
  __bf16* yp = Y + (size_t)m0 * (HDIM*BASES) + col;
  if (fullTile) {       // uniform branch: unguarded 8-store fast path
    #pragma unroll
    for (int g = 0; g < 8; ++g)
      yp[(size_t)g * (HDIM*BASES)] = (__bf16)c[g];
  } else {
    #pragma unroll
    for (int g = 0; g < 8; ++g)
      if (m0 + g < nrows) yp[(size_t)g * (HDIM*BASES)] = (__bf16)c[g];
  }
}

// ---------------- attention pooling ----------------
__global__ void k_logits(const float* __restrict__ h, const float* __restrict__ gW,
                         const float* __restrict__ gb, float* __restrict__ logits, int N)
{
  int wid = threadIdx.x >> 5, lane = threadIdx.x & 31;
  int n = blockIdx.x * 8 + wid;
  if (n >= N) return;
  const float* row = h + (long long)n * HDIM;
  float a = 0.f;
  #pragma unroll
  for (int j = 0; j < 4; ++j) { int c = lane + j*32; a += row[c] * gW[c]; }
  #pragma unroll
  for (int m = 16; m >= 1; m >>= 1) a += __shfl_xor(a, m, 32);
  if (lane == 0) logits[n] = a + gb[0];
}

__global__ void k_softmax_stats(const float* __restrict__ logits, int N,
                                float* __restrict__ stats, float* __restrict__ out)
{
  __shared__ float sm[256];
  int t = threadIdx.x;
  float mx = -3.402823e38f;
  for (int i = t; i < N; i += 256) mx = fmaxf(mx, logits[i]);
  sm[t] = mx; __syncthreads();
  for (int s = 128; s >= 1; s >>= 1) { if (t < s) sm[t] = fmaxf(sm[t], sm[t+s]); __syncthreads(); }
  float gmax = sm[0]; __syncthreads();
  float z = 0.f;
  for (int i = t; i < N; i += 256) z += __expf(logits[i] - gmax);
  sm[t] = z; __syncthreads();
  for (int s = 128; s >= 1; s >>= 1) { if (t < s) sm[t] += sm[t+s]; __syncthreads(); }
  if (t == 0) { stats[0] = gmax; stats[1] = 1.f / sm[0]; }
  if (t < HDIM) out[t] = 0.f;   // zero d_out before pooling atomics
}

__global__ void k_softmax_w(float* __restrict__ logits, const float* __restrict__ stats, int N)
{
  int i = blockIdx.x * blockDim.x + threadIdx.x;
  if (i < N) logits[i] = __expf(logits[i] - stats[0]) * stats[1];
}

// 128 threads = 128 channels; each block strides over nodes -> 16K total atomics
__global__ void k_pool(const float* __restrict__ h, const float* __restrict__ w,
                       int N, float* __restrict__ out)
{
  int c = threadIdx.x;
  float acc = 0.f;
  for (int n = blockIdx.x; n < N; n += gridDim.x)
    acc += w[n] * h[(long long)n * HDIM + c];
  atomicAdd(&out[c], acc);
}

// ---------------- launch ----------------
extern "C" void kernel_launch(void* const* d_in, const int* in_sizes, int n_in,
                              void* d_out, int out_size, void* d_ws, size_t ws_size,
                              hipStream_t stream)
{
  const int*   x        = (const int*)  d_in[0];
  const int*   src      = (const int*)  d_in[1];
  const int*   dst      = (const int*)  d_in[2];
  const int*   rel      = (const int*)  d_in[3];
  const float* norm     = (const float*)d_in[4];
  const float* V_in     = (const float*)d_in[5];
  const float* comp_in  = (const float*)d_in[6];
  const float* bias_in  = (const float*)d_in[7];
  const float* V_h      = (const float*)d_in[8];
  const float* comp_h   = (const float*)d_in[9];
  const float* bias_h   = (const float*)d_in[10];
  const float* V_out    = (const float*)d_in[11];
  const float* comp_out = (const float*)d_in[12];
  const float* bias_out = (const float*)d_in[13];
  const float* gate_W   = (const float*)d_in[14];
  const float* gate_b   = (const float*)d_in[15];
  float* out = (float*)d_out;

  const int N = in_sizes[0];
  const int E = in_sizes[1];

  char* wsb = (char*)d_ws;
  size_t off = 0;
  auto alloc = [&](size_t bytes) -> char* {
    char* p = wsb + off;
    off = (off + bytes + 255) & ~(size_t)255;
    return p;
  };
  float*  h      = (float*) alloc((size_t)N * HDIM * 4);
  __bf16* hbf    = (__bf16*)alloc((size_t)N * HDIM * 2);
  __bf16* Ybf    = (__bf16*)alloc((size_t)N * HDIM * BASES * 2);
  __bf16* Wpk2   = (__bf16*)alloc((size_t)HDIM * HDIM * BASES * 2);
  __bf16* Wpk3   = (__bf16*)alloc((size_t)HDIM * HDIM * BASES * 2);
  float*  logits = (float*) alloc((size_t)N * 4);
  float*  stats  = (float*) alloc(256);
  (void)ws_size; (void)n_in; (void)out_size;

  const int totalH   = N * HDIM;
  const dim3 blk(256);
  const int zb  = (totalH/4 + 255) / 256;
  const int fb  = (totalH   + 255) / 256;
  const int eb  = (E + 7) / 8;
  const int rt  = (N + 15) / 16;
  const int pkT = HDIM * HDIM * BASES;
  const int pkB = (pkT + 255) / 256;

  // ---- layer 1 (featureless: gather V_in rows) ----
  k_zero_f32<<<zb, blk, 0, stream>>>(h, totalH/4);
  k_edge_combine<float><<<eb, blk, 0, stream>>>(src, dst, rel, norm, comp_in,
      V_in, (long long)HDIM, (long long)N * HDIM, x, h, E);
  k_finalize<<<fb, blk, 0, stream>>>(h, bias_in, hbf, totalH, 1);

  // ---- layer 2 ----
  k_pack_w<<<pkB, blk, 0, stream>>>(V_h, Wpk2, pkT);
  k_gemm_wmma<<<dim3(rt, 4), blk, 0, stream>>>(hbf, Wpk2, Ybf, N);
  k_zero_f32<<<zb, blk, 0, stream>>>(h, totalH/4);
  k_edge_combine<__bf16><<<eb, blk, 0, stream>>>(src, dst, rel, norm, comp_h,
      Ybf, (long long)(HDIM*BASES), (long long)HDIM, nullptr, h, E);
  k_finalize<<<fb, blk, 0, stream>>>(h, bias_h, hbf, totalH, 1);

  // ---- layer 3 ----
  k_pack_w<<<pkB, blk, 0, stream>>>(V_out, Wpk3, pkT);
  k_gemm_wmma<<<dim3(rt, 4), blk, 0, stream>>>(hbf, Wpk3, Ybf, N);
  k_zero_f32<<<zb, blk, 0, stream>>>(h, totalH/4);
  k_edge_combine<__bf16><<<eb, blk, 0, stream>>>(src, dst, rel, norm, comp_out,
      Ybf, (long long)(HDIM*BASES), (long long)HDIM, nullptr, h, E);
  k_finalize<<<fb, blk, 0, stream>>>(h, bias_out, (__bf16*)nullptr, totalH, 0);

  // ---- global attention pooling ----
  k_logits<<<(N + 7)/8, blk, 0, stream>>>(h, gate_W, gate_b, logits, N);
  k_softmax_stats<<<1, blk, 0, stream>>>(logits, N, stats, out);
  k_softmax_w<<<(N + 255)/256, blk, 0, stream>>>(logits, stats, N);
  k_pool<<<128, 128, 0, stream>>>(h, logits, N, out);
}
